// FusedLlamaLowBitMultiDecoderlayer_34076270526685
// MI455X (gfx1250) — matrix operand order repaired and checked by reading, not compiled, not run on hardware
//
#include <hip/hip_runtime.h>
#include <hip/hip_bf16.h>
#include <stdint.h>

#define S_LEN 2048
#define HID   2048
#define NH    32
#define NKV   8
#define HD    64
#define I_FF  5632
#define L_NUM 4

typedef __attribute__((ext_vector_type(16))) __bf16 v16bf;
typedef __attribute__((ext_vector_type(8)))  __bf16 v8bf;
typedef __attribute__((ext_vector_type(8)))  float  v8f;

static __device__ inline uint16_t f2bf(float f) {
    uint32_t u = __float_as_uint(f);
    u += 0x7FFFu + ((u >> 16) & 1u);   // round-to-nearest-even
    return (uint16_t)(u >> 16);
}

static __device__ inline v16bf cat16(v8bf lo, v8bf hi) {
    v16bf r;
#pragma unroll
    for (int i = 0; i < 8; ++i) { r[i] = lo[i]; r[i + 8] = hi[i]; }
    return r;
}

static __device__ inline float rmax16(float v) {
#pragma unroll
    for (int o = 8; o; o >>= 1) v = fmaxf(v, __shfl_xor(v, o, 16));
    return v;
}
static __device__ inline float rsum16(float v) {
#pragma unroll
    for (int o = 8; o; o >>= 1) v += __shfl_xor(v, o, 16);
    return v;
}

// Async global -> LDS copy, 16B per lane (ASYNCcnt-tracked, CDNA5).
// lds_off = LDS byte offset (low 32 bits of generic shared pointer, ISA 10.2).
static __device__ inline void async_ld_b128(uint32_t lds_off, const void* g) {
    asm volatile("global_load_async_to_lds_b128 %0, %1, off"
                 :: "v"(lds_off), "v"(g) : "memory");
}
static __device__ inline void wait_async_le3() {
    asm volatile("s_wait_asynccnt 0x3" ::: "memory");
}
static __device__ inline void wait_async_0() {
    asm volatile("s_wait_asynccnt 0x0" ::: "memory");
}

// ---------------------------------------------------------------------------
// fp32 -> bf16 bulk conversion (weights, once per call; memory bound)
// ---------------------------------------------------------------------------
__global__ __launch_bounds__(256) void f32_to_bf16_kernel(
    const float* __restrict__ a, uint16_t* __restrict__ b, int n)
{
    int i = (blockIdx.x * 256 + threadIdx.x) * 4;
    if (i < n) {
        float4 v = *(const float4*)(a + i);
        ushort4 o;
        o.x = f2bf(v.x); o.y = f2bf(v.y); o.z = f2bf(v.z); o.w = f2bf(v.w);
        *(ushort4*)(b + i) = o;
    }
}

// ---------------------------------------------------------------------------
// RMSNorm -> bf16
// ---------------------------------------------------------------------------
__global__ __launch_bounds__(256) void rmsnorm_bf16_kernel(
    const float* __restrict__ h, const float* __restrict__ w,
    uint16_t* __restrict__ out, int Hn)
{
    __shared__ float red[256];
    int s = blockIdx.x, t = threadIdx.x;
    const float* row = h + (size_t)s * Hn;
    float acc = 0.f;
    for (int i = t; i < Hn; i += 256) { float v = row[i]; acc += v * v; }
    red[t] = acc;
    __syncthreads();
    for (int st = 128; st; st >>= 1) { if (t < st) red[t] += red[t + st]; __syncthreads(); }
    float scale = rsqrtf(red[0] / (float)Hn + 1e-5f);
    for (int i = t; i < Hn; i += 256)
        out[(size_t)s * Hn + i] = f2bf(row[i] * scale * w[i]);
}

// ---------------------------------------------------------------------------
// GEMM: C[M,N] = A[M,K](bf16) * W[N,K](bf16)^T (+ optional residual)
// Block tile 128x64, 8 waves (4M x 2N), wave tile 32x32 = 4 WMMA per K-step.
// Double-buffered async global->LDS staging (ASYNCcnt), K-step 32, with the
// final iteration peeled so the steady-state loop is branch-free.
// Requires M%128==0, N%64==0, K%32==0, K/32 >= 2.
// ---------------------------------------------------------------------------
__global__ __launch_bounds__(256) void gemm_bf16_wmma_kernel(
    const uint16_t* __restrict__ A, const uint16_t* __restrict__ W,
    float* __restrict__ C, const float* __restrict__ R,
    int M, int N, int K)
{
    __shared__ __align__(16) uint16_t At[2][128 * 32];   // 2 x 8 KB
    __shared__ __align__(16) uint16_t Wt[2][64 * 32];    // 2 x 4 KB

    int tid  = threadIdx.x;
    int wave = tid >> 5, lane = tid & 31;
    int wm = wave >> 1, wn = wave & 1;
    int mrow = lane & 15, khalf = lane >> 4;
    int m0 = blockIdx.y * 128, n0 = blockIdx.x * 64;

    v8f acc[2][2];
#pragma unroll
    for (int am = 0; am < 2; ++am)
#pragma unroll
        for (int bn = 0; bn < 2; ++bn)
#pragma unroll
            for (int i = 0; i < 8; ++i) acc[am][bn][i] = 0.f;

    int lrA = tid >> 1;           // 0..127
    int lcA = (tid & 1) * 16;     // 0 or 16
    int lrW = tid >> 2;           // 0..63
    int lcW = (tid & 3) * 8;      // 0,8,16,24
    const uint16_t* gA = A + (size_t)(m0 + lrA) * K + lcA;
    const uint16_t* gW = W + (size_t)(n0 + lrW) * K + lcW;

    // LDS byte offsets for this thread's async destinations (buffer 0)
    uint32_t ldsA = (uint32_t)(uintptr_t)(&At[0][lrA * 32 + lcA]);
    uint32_t ldsW = (uint32_t)(uintptr_t)(&Wt[0][lrW * 32 + lcW]);
    const uint32_t strA = (uint32_t)sizeof(uint16_t) * 128 * 32;  // 8192
    const uint32_t strW = (uint32_t)sizeof(uint16_t) * 64 * 32;   // 4096

    // wave-level WMMA step on buffer `cur`
    auto compute = [&](int cur) {
        // A vectors: rows wm*32 + {0,16} + mrow; K chunks (khalf*8, +16)
        const uint16_t* ar0 = &At[cur][(wm * 32 + mrow) * 32 + khalf * 8];
        const uint16_t* ar1 = ar0 + 16 * 32;
        v16bf a0 = cat16(*(const v8bf*)ar0, *(const v8bf*)(ar0 + 16));
        v16bf a1 = cat16(*(const v8bf*)ar1, *(const v8bf*)(ar1 + 16));
        // B vectors: cols wn*32 + {0,16} + mrow; K chunks (khalf*8, +16)
        const uint16_t* bp0 = &Wt[cur][(wn * 32 + mrow) * 32 + khalf * 8];
        const uint16_t* bp1 = bp0 + 16 * 32;
        v16bf b0 = cat16(*(const v8bf*)bp0, *(const v8bf*)(bp0 + 16));
        v16bf b1 = cat16(*(const v8bf*)bp1, *(const v8bf*)(bp1 + 16));

        acc[0][0] = __builtin_amdgcn_wmma_f32_16x16x32_bf16(false, a0, false, b0,
                                                            (short)0, acc[0][0], false, false);
        acc[0][1] = __builtin_amdgcn_wmma_f32_16x16x32_bf16(false, a0, false, b1,
                                                            (short)0, acc[0][1], false, false);
        acc[1][0] = __builtin_amdgcn_wmma_f32_16x16x32_bf16(false, a1, false, b0,
                                                            (short)0, acc[1][0], false, false);
        acc[1][1] = __builtin_amdgcn_wmma_f32_16x16x32_bf16(false, a1, false, b1,
                                                            (short)0, acc[1][1], false, false);
    };

    // prologue: stage tile 0
    async_ld_b128(ldsA,      gA);
    async_ld_b128(ldsA + 16, gA + 8);
    async_ld_b128(ldsW,      gW);

    const int nsteps = K >> 5;
    // steady state: always stage the next tile while consuming the current one
    for (int it = 0; it < nsteps - 1; ++it) {
        int cur = it & 1, nxt = cur ^ 1;
        const uint16_t* gA2 = gA + ((it + 1) << 5);
        const uint16_t* gW2 = gW + ((it + 1) << 5);
        async_ld_b128(ldsA + nxt * strA,      gA2);
        async_ld_b128(ldsA + nxt * strA + 16, gA2 + 8);
        async_ld_b128(ldsW + nxt * strW,      gW2);
        // speculative prefetch ~7 K-steps ahead (dropped silently if unmapped)
        __builtin_prefetch(gA2 + 224, 0, 1);   // global_prefetch_b8
        __builtin_prefetch(gW2 + 224, 0, 1);
        wait_async_le3();   // in-order completion: current tile's 3 loads landed
        __syncthreads();
        compute(cur);
        __syncthreads();    // all waves done reading `cur` before it is re-staged
    }
    // peeled last iteration
    {
        wait_async_0();
        __syncthreads();
        compute((nsteps - 1) & 1);
    }

#pragma unroll
    for (int am = 0; am < 2; ++am) {
        int crow = m0 + wm * 32 + am * 16 + khalf * 8;
#pragma unroll
        for (int bn = 0; bn < 2; ++bn) {
            int ccol = n0 + wn * 32 + bn * 16 + mrow;
#pragma unroll
            for (int r = 0; r < 8; ++r) {
                size_t i0 = (size_t)(crow + r) * N + ccol;
                float v = acc[am][bn][r];
                if (R) v += R[i0];
                C[i0] = v;
            }
        }
    }
}

// ---------------------------------------------------------------------------
// RoPE: rotate Q,K; emit bf16 [head][s][d] tensors for attention and the fp32
// K/V caches into d_out. Q pre-scaled by 1/sqrt(HD).
// ---------------------------------------------------------------------------
__global__ __launch_bounds__(256) void rope_qkv_kernel(
    const float* __restrict__ qf, const float* __restrict__ kf,
    const float* __restrict__ vf, const float* __restrict__ cosT,
    const float* __restrict__ sinT, const int* __restrict__ pos,
    uint16_t* __restrict__ qb, uint16_t* __restrict__ kb,
    uint16_t* __restrict__ vb, float* __restrict__ kout,
    float* __restrict__ vout)
{
    int s = blockIdx.x, t = threadIdx.x;
    int p = pos[s];
    const float* cr = cosT + (size_t)p * HD;
    const float* sr = sinT + (size_t)p * HD;

    for (int i = t; i < NH * HD; i += 256) {
        int head = i >> 6, d = i & 63;
        float v = qf[(size_t)s * NH * HD + i];
        float o = qf[(size_t)s * NH * HD + (head << 6) + ((d < 32) ? d + 32 : d - 32)];
        float r = (d < 32) ? (v * cr[d] - o * sr[d]) : (v * cr[d] + o * sr[d]);
        qb[((size_t)head * S_LEN + s) * HD + d] = f2bf(r * 0.125f);  // 1/sqrt(64)
    }
    for (int i = t; i < NKV * HD; i += 256) {
        int head = i >> 6, d = i & 63;
        size_t dst = ((size_t)head * S_LEN + s) * HD + d;
        float v = kf[(size_t)s * NKV * HD + i];
        float o = kf[(size_t)s * NKV * HD + (head << 6) + ((d < 32) ? d + 32 : d - 32)];
        float r = (d < 32) ? (v * cr[d] - o * sr[d]) : (v * cr[d] + o * sr[d]);
        kb[dst] = f2bf(r);
        kout[dst] = r;
        float vv = vf[(size_t)s * NKV * HD + i];
        vb[dst] = f2bf(vv);
        vout[dst] = vv;
    }
}

// ---------------------------------------------------------------------------
// Causal flash attention with WMMA (bf16 inputs, fp32 accumulation).
// grid = (NH, S/128); block = 256 (8 waves, one 16-query tile per wave).
// ---------------------------------------------------------------------------
__global__ __launch_bounds__(256) void flash_attn_wmma_kernel(
    const uint16_t* __restrict__ qb, const uint16_t* __restrict__ kb,
    const uint16_t* __restrict__ vb, uint16_t* __restrict__ attn_out)
{
    __shared__ __align__(16) uint16_t ktile[32 * 64];   // [k][d]
    __shared__ __align__(16) uint16_t vtile[64 * 32];   // [d][k]  (transposed)
    __shared__ __align__(16) uint16_t ptile[8 * 16 * 32];

    int tid  = threadIdx.x;
    int wave = tid >> 5, lane = tid & 31;
    int mrow = lane & 15, khalf = lane >> 4;
    int h = blockIdx.x, kvh = h >> 2;     // GQA: 4 Q heads per KV head
    int qrow0 = blockIdx.y * 128 + wave * 16;

    const uint16_t* qr = qb + ((size_t)h * S_LEN + qrow0 + mrow) * HD;
    v16bf qa0 = cat16(*(const v8bf*)(qr + khalf * 8),
                      *(const v8bf*)(qr + khalf * 8 + 16));
    v16bf qa1 = cat16(*(const v8bf*)(qr + 32 + khalf * 8),
                      *(const v8bf*)(qr + 32 + khalf * 8 + 16));

    float m_run[8], l_run[8];
#pragma unroll
    for (int r = 0; r < 8; ++r) { m_run[r] = -1e30f; l_run[r] = 0.f; }
    v8f oacc[4];
#pragma unroll
    for (int t2 = 0; t2 < 4; ++t2)
#pragma unroll
        for (int i = 0; i < 8; ++i) oacc[t2][i] = 0.f;

    int kend = blockIdx.y * 128 + 127;
    int kr = tid >> 3;            // 0..31
    int d0 = (tid & 7) * 8;       // 0..56
    uint16_t* pt = ptile + wave * (16 * 32);

    for (int kb0 = 0; kb0 <= kend; kb0 += 32) {
        __syncthreads();
        *(uint4*)(ktile + kr * 64 + d0) =
            *(const uint4*)(kb + ((size_t)kvh * S_LEN + kb0 + kr) * HD + d0);
        union { uint4 u; uint16_t hx[8]; } vv;
        vv.u = *(const uint4*)(vb + ((size_t)kvh * S_LEN + kb0 + kr) * HD + d0);
#pragma unroll
        for (int j = 0; j < 8; ++j) vtile[(d0 + j) * 32 + kr] = vv.hx[j];
        __syncthreads();

        // ---- scores: S = Q * K^T  (two 16-key subtiles) ----
        v8f s0, s1;
#pragma unroll
        for (int i = 0; i < 8; ++i) { s0[i] = 0.f; s1[i] = 0.f; }
        {
            const uint16_t* kp = ktile + mrow * 64 + khalf * 8;
            v16bf b0 = cat16(*(const v8bf*)kp, *(const v8bf*)(kp + 16));
            v16bf b1 = cat16(*(const v8bf*)(kp + 32), *(const v8bf*)(kp + 48));
            s0 = __builtin_amdgcn_wmma_f32_16x16x32_bf16(false, qa0, false, b0,
                                                         (short)0, s0, false, false);
            s0 = __builtin_amdgcn_wmma_f32_16x16x32_bf16(false, qa1, false, b1,
                                                         (short)0, s0, false, false);
        }
        {
            const uint16_t* kp = ktile + (16 + mrow) * 64 + khalf * 8;
            v16bf b0 = cat16(*(const v8bf*)kp, *(const v8bf*)(kp + 16));
            v16bf b1 = cat16(*(const v8bf*)(kp + 32), *(const v8bf*)(kp + 48));
            s1 = __builtin_amdgcn_wmma_f32_16x16x32_bf16(false, qa0, false, b0,
                                                         (short)0, s1, false, false);
            s1 = __builtin_amdgcn_wmma_f32_16x16x32_bf16(false, qa1, false, b1,
                                                         (short)0, s1, false, false);
        }

        // ---- mask + online softmax ----
#pragma unroll
        for (int r = 0; r < 8; ++r) {
            int qidx = qrow0 + khalf * 8 + r;
            float v0 = s0[r] + ((kb0 + mrow)      > qidx ? -1e4f : 0.f);
            float v1 = s1[r] + ((kb0 + 16 + mrow) > qidx ? -1e4f : 0.f);
            float mx = rmax16(fmaxf(v0, v1));
            float mn = fmaxf(m_run[r], mx);
            float p0 = __expf(v0 - mn), p1 = __expf(v1 - mn);
            float sum = rsum16(p0 + p1);
            float fac = __expf(m_run[r] - mn);
            l_run[r] = l_run[r] * fac + sum;
            m_run[r] = mn;
            oacc[0][r] *= fac; oacc[1][r] *= fac;
            oacc[2][r] *= fac; oacc[3][r] *= fac;
            uint16_t* pw = pt + (khalf * 8 + r) * 32 + mrow;
            pw[0]  = f2bf(p0);
            pw[16] = f2bf(p1);
        }

        // ---- O += P * V ----
        const uint16_t* pr = pt + mrow * 32 + khalf * 8;
        v16bf pa = cat16(*(const v8bf*)pr, *(const v8bf*)(pr + 16));
#pragma unroll
        for (int t2 = 0; t2 < 4; ++t2) {
            const uint16_t* vp = vtile + (t2 * 16 + mrow) * 32 + khalf * 8;
            v16bf bv = cat16(*(const v8bf*)vp, *(const v8bf*)(vp + 16));
            oacc[t2] = __builtin_amdgcn_wmma_f32_16x16x32_bf16(
                false, pa, false, bv, (short)0, oacc[t2], false, false);
        }
    }

#pragma unroll
    for (int r = 0; r < 8; ++r) {
        float inv = 1.f / l_run[r];
        int row = qrow0 + khalf * 8 + r;
#pragma unroll
        for (int t2 = 0; t2 < 4; ++t2)
            attn_out[(size_t)row * (NH * HD) + h * HD + t2 * 16 + mrow] =
                f2bf(oacc[t2][r] * inv);
    }
}

// ---------------------------------------------------------------------------
// SiLU(g) * u -> bf16
// ---------------------------------------------------------------------------
__global__ __launch_bounds__(256) void silu_mul_kernel(
    const float* __restrict__ g, const float* __restrict__ u,
    uint16_t* __restrict__ m, int n)
{
    int i = blockIdx.x * 256 + threadIdx.x;
    if (i < n) {
        float gv = g[i];
        float sv = gv / (1.f + __expf(-gv));
        m[i] = f2bf(sv * u[i]);
    }
}

__global__ __launch_bounds__(256) void copy_f32_kernel(
    const float* __restrict__ a, float* __restrict__ b, int n)
{
    int i = blockIdx.x * 256 + threadIdx.x;
    if (i < n) b[i] = a[i];
}

// ---------------------------------------------------------------------------
extern "C" void kernel_launch(void* const* d_in, const int* in_sizes, int n_in,
                              void* d_out, int out_size, void* d_ws, size_t ws_size,
                              hipStream_t stream) {
    (void)in_sizes; (void)n_in; (void)out_size; (void)ws_size;
    const float* x      = (const float*)d_in[0];
    const int*   pos    = (const int*)d_in[2];
    const float* cosT   = (const float*)d_in[3];
    const float* sinT   = (const float*)d_in[4];
    const float* ln1_w  = (const float*)d_in[5];
    const float* ln2_w  = (const float*)d_in[6];
    const float* q_w    = (const float*)d_in[7];
    const float* k_w    = (const float*)d_in[8];
    const float* v_w    = (const float*)d_in[9];
    const float* o_w    = (const float*)d_in[10];
    const float* gate_w = (const float*)d_in[11];
    const float* up_w   = (const float*)d_in[12];
    const float* down_w = (const float*)d_in[13];

    float* out_h = (float*)d_out;
    float* out_k = out_h + (size_t)S_LEN * HID;
    float* out_v = out_k + (size_t)L_NUM * NKV * S_LEN * HD;

    char* ws = (char*)d_ws;
    size_t off = 0;
    auto carve = [&](size_t bytes) -> char* {
        char* p = ws + off;
        off = (off + bytes + 255) & ~(size_t)255;
        return p;
    };

    const size_t nQ = (size_t)L_NUM * (NH * HD) * HID;
    const size_t nK = (size_t)L_NUM * (NKV * HD) * HID;
    const size_t nO = (size_t)L_NUM * HID * HID;
    const size_t nG = (size_t)L_NUM * I_FF * HID;
    uint16_t* qw16 = (uint16_t*)carve(nQ * 2);
    uint16_t* kw16 = (uint16_t*)carve(nK * 2);
    uint16_t* vw16 = (uint16_t*)carve(nK * 2);
    uint16_t* ow16 = (uint16_t*)carve(nO * 2);
    uint16_t* gw16 = (uint16_t*)carve(nG * 2);
    uint16_t* uw16 = (uint16_t*)carve(nG * 2);
    uint16_t* dw16 = (uint16_t*)carve(nG * 2);

    float*    hbuf   = (float*)   carve((size_t)S_LEN * HID * 4);
    uint16_t* hn16   = (uint16_t*)carve((size_t)S_LEN * HID * 2);
    float*    qf     = (float*)   carve((size_t)S_LEN * NH * HD * 4);
    float*    kf     = (float*)   carve((size_t)S_LEN * NKV * HD * 4);
    float*    vf     = (float*)   carve((size_t)S_LEN * NKV * HD * 4);
    uint16_t* qb     = (uint16_t*)carve((size_t)NH * S_LEN * HD * 2);
    uint16_t* kb16   = (uint16_t*)carve((size_t)NKV * S_LEN * HD * 2);
    uint16_t* vb16   = (uint16_t*)carve((size_t)NKV * S_LEN * HD * 2);
    uint16_t* attn16 = (uint16_t*)carve((size_t)S_LEN * HID * 2);
    float*    gbuf   = (float*)   carve((size_t)S_LEN * I_FF * 4);
    float*    ubuf   = (float*)   carve((size_t)S_LEN * I_FF * 4);
    uint16_t* m16    = (uint16_t*)carve((size_t)S_LEN * I_FF * 2);

    auto cvt = [&](const float* src, uint16_t* dst, size_t n) {
        f32_to_bf16_kernel<<<(unsigned)((n / 4 + 255) / 256), 256, 0, stream>>>(src, dst, (int)n);
    };
    cvt(q_w, qw16, nQ);  cvt(k_w, kw16, nK);  cvt(v_w, vw16, nK);
    cvt(o_w, ow16, nO);  cvt(gate_w, gw16, nG); cvt(up_w, uw16, nG);
    cvt(down_w, dw16, nG);

    const int nSH = S_LEN * HID;
    copy_f32_kernel<<<(nSH + 255) / 256, 256, 0, stream>>>(x, hbuf, nSH);

    for (int l = 0; l < L_NUM; ++l) {
        const uint16_t* qw = qw16 + (size_t)l * (NH * HD) * HID;
        const uint16_t* kw = kw16 + (size_t)l * (NKV * HD) * HID;
        const uint16_t* vw = vw16 + (size_t)l * (NKV * HD) * HID;
        const uint16_t* ow = ow16 + (size_t)l * HID * HID;
        const uint16_t* gw = gw16 + (size_t)l * I_FF * HID;
        const uint16_t* uw = uw16 + (size_t)l * I_FF * HID;
        const uint16_t* dw = dw16 + (size_t)l * HID * I_FF;

        rmsnorm_bf16_kernel<<<S_LEN, 256, 0, stream>>>(hbuf, ln1_w + (size_t)l * HID, hn16, HID);

        gemm_bf16_wmma_kernel<<<dim3((NH * HD) / 64, S_LEN / 128), 256, 0, stream>>>(
            hn16, qw, qf, nullptr, S_LEN, NH * HD, HID);
        gemm_bf16_wmma_kernel<<<dim3((NKV * HD) / 64, S_LEN / 128), 256, 0, stream>>>(
            hn16, kw, kf, nullptr, S_LEN, NKV * HD, HID);
        gemm_bf16_wmma_kernel<<<dim3((NKV * HD) / 64, S_LEN / 128), 256, 0, stream>>>(
            hn16, vw, vf, nullptr, S_LEN, NKV * HD, HID);

        rope_qkv_kernel<<<S_LEN, 256, 0, stream>>>(
            qf, kf, vf, cosT, sinT, pos, qb, kb16, vb16,
            out_k + (size_t)l * NKV * S_LEN * HD,
            out_v + (size_t)l * NKV * S_LEN * HD);

        flash_attn_wmma_kernel<<<dim3(NH, S_LEN / 128), 256, 0, stream>>>(
            qb, kb16, vb16, attn16);

        gemm_bf16_wmma_kernel<<<dim3(HID / 64, S_LEN / 128), 256, 0, stream>>>(
            attn16, ow, hbuf, hbuf, S_LEN, HID, HID);   // fused residual, in-place

        rmsnorm_bf16_kernel<<<S_LEN, 256, 0, stream>>>(hbuf, ln2_w + (size_t)l * HID, hn16, HID);

        gemm_bf16_wmma_kernel<<<dim3(I_FF / 64, S_LEN / 128), 256, 0, stream>>>(
            hn16, gw, gbuf, nullptr, S_LEN, I_FF, HID);
        gemm_bf16_wmma_kernel<<<dim3(I_FF / 64, S_LEN / 128), 256, 0, stream>>>(
            hn16, uw, ubuf, nullptr, S_LEN, I_FF, HID);

        const int nSI = S_LEN * I_FF;
        silu_mul_kernel<<<(nSI + 255) / 256, 256, 0, stream>>>(gbuf, ubuf, m16, nSI);

        gemm_bf16_wmma_kernel<<<dim3(HID / 64, S_LEN / 128), 256, 0, stream>>>(
            m16, dw, hbuf, hbuf, S_LEN, HID, I_FF);     // fused residual, in-place
    }

    copy_f32_kernel<<<(nSH + 255) / 256, 256, 0, stream>>>(hbuf, out_h, nSH);
}
